// GlobalInteraction_24764781429080
// MI455X (gfx1250) — matrix operand order repaired and checked by reading, not compiled, not run on hardware
//
// MI455X (gfx1250) fused kernel for per-expert attention + deep MLP + LayerNorm.
//
// Roofline: mandatory HBM traffic ~0.55 GB -> ~24 us @ 23.3 TB/s. Matrix math
// ~73 GFLOP -> ~27 us at dense-bf16 WMMA rate, so bf16 (f32 accumulate) is the
// balanced precision; fp32 WMMA (16x16x4) would be ~9x slower than the memory
// floor. Everything is fused into one kernel per (16-batch, expert) tile so X
// is read once and out written once. Weights are pre-packed to bf16 (16 MB in
// d_ws) by a tiny one-shot kernel.
//
// Requires ws_size >= 16 MiB.

#include <hip/hip_runtime.h>
#include <stdint.h>

#define B_TOT   4096
#define NG      64
#define DM      256
#define NE      4
#define TPE     16
#define FF      (TPE * DM)   // 4096 = flattened partition
#define TB      16           // batches per workgroup (== WMMA M)
#define NTHREADS 256         // 8 wave32 waves
#define EPS     1e-5f

typedef __bf16 bf16x16 __attribute__((ext_vector_type(16)));
typedef float  f32x8   __attribute__((ext_vector_type(8)));

union FragAB {
    bf16x16  bf;
    uint32_t u[8];
    uint4    q[2];
};
static_assert(sizeof(FragAB) == 32, "frag must be 8 VGPRs");

__device__ __forceinline__ uint16_t f2bf(float f) {
    // round-to-nearest-even float -> bf16
    uint32_t u = __float_as_uint(f);
    uint32_t r = u + 0x7FFFu + ((u >> 16) & 1u);
    return (uint16_t)(r >> 16);
}
__device__ __forceinline__ float bf2f(uint16_t h) {
    return __uint_as_float(((uint32_t)h) << 16);
}
__device__ __forceinline__ uint32_t pack2bf(float lo, float hi) {
    return (uint32_t)f2bf(lo) | ((uint32_t)f2bf(hi) << 16);
}

__device__ __forceinline__ f32x8 wmma_bf16(const FragAB& a, const FragAB& b, f32x8 c) {
    // (neg_a, A, neg_b, B, c_mod, C, reuse_a, reuse_b)
    return __builtin_amdgcn_wmma_f32_16x16x32_bf16(false, a.bf, false, b.bf,
                                                   (short)0, c, false, false);
}

// ---------------------------------------------------------------------------
// One-shot weight repack: f32 row-major W -> bf16, K-pairs packed in a dword,
// column(n)-major so WMMA B-fragments become contiguous uint4 loads.
//   P1[(e*256 + n)*2048 + k/2]  from W1[e][k][n]   (K=4096, N=256)
//   P2[(e*4096 + n)*128  + k/2] from W2[e][k][n]   (K=256,  N=4096)
// ---------------------------------------------------------------------------
__global__ void pack_weights_kernel(const float* __restrict__ W1,
                                    const float* __restrict__ W2,
                                    uint32_t* __restrict__ P1,
                                    uint32_t* __restrict__ P2) {
    const int n1 = NE * DM * (FF / 2);           // 2M dwords
    int idx = blockIdx.x * blockDim.x + threadIdx.x;
    if (idx < n1) {
        int k2 = idx & (FF / 2 - 1);             // 0..2047
        int n  = (idx >> 11) & (DM - 1);         // 0..255
        int e  = idx >> 19;                      // 0..3
        const float* w = W1 + ((size_t)e * FF + (size_t)(2 * k2)) * DM + n;
        P1[idx] = pack2bf(w[0], w[DM]);
    } else if (idx < 2 * n1) {
        int j  = idx - n1;
        int k2 = j & (DM / 2 - 1);               // 0..127
        int n  = (j >> 7) & (FF - 1);            // 0..4095
        int e  = j >> 19;                        // 0..3
        const float* w = W2 + ((size_t)e * DM + (size_t)(2 * k2)) * FF + n;
        P2[j] = pack2bf(w[0], w[FF]);
    }
}

// ---------------------------------------------------------------------------
// Fused kernel. Grid (B/16, E); block 256 threads = 8 waves.
// LDS: Xs 128KB (bf16 X tile; becomes X_dot+Xp in phase C), Hs 8KB, Ps 8KB,
//      b1/gamma/beta 3KB  -> ~147KB (fits the 160KB CU-mode budget).
// ---------------------------------------------------------------------------
__global__ void __launch_bounds__(NTHREADS)
fused_moe_kernel(const float* __restrict__ X,
                 const float* __restrict__ b1g,
                 const float* __restrict__ b2g,
                 const float* __restrict__ gammag,
                 const float* __restrict__ betag,
                 const uint32_t* __restrict__ P1,
                 const uint32_t* __restrict__ P2,
                 float* __restrict__ out) {
    __shared__ uint16_t Xs[TB * FF];            // [b][t*256+dd] bf16; later Ypart
    __shared__ uint16_t Hs[TB * DM];            // h = relu(Xf@W1+b1), bf16
    __shared__ uint16_t Ps[TB * TPE * TPE];     // softmax probs, bf16
    __shared__ float    b1s[DM], gs[DM], bs[DM];

    const int tid   = threadIdx.x;
    const int lane  = tid & 31;
    const int w     = tid >> 5;                 // wave 0..7
    const int ln16  = lane & 15;
    const int hi    = lane >> 4;                // lane half (0/1)
    const int btile = blockIdx.x;               // 0..255
    const int e     = blockIdx.y;               // 0..3

    // ---- Phase A: load X tile f32 -> bf16 LDS; load per-expert vectors ----
    {
        const int base0 = btile * (TB * NG * DM) + e * (TPE * DM);
        for (int i = 0; i < 64; ++i) {
            int flat = (i * NTHREADS + tid) * 4;          // element index in tile
            int b    = flat >> 12;                        // /4096
            int rem  = flat & (FF - 1);
            float4 v = *(const float4*)(X + (size_t)base0 + (size_t)b * (NG * DM) + rem);
            uint2 p;
            p.x = pack2bf(v.x, v.y);
            p.y = pack2bf(v.z, v.w);
            *(uint2*)(&Xs[flat]) = p;
        }
        b1s[tid] = b1g[e * DM + tid];
        gs[tid]  = gammag[e * DM + tid];
        bs[tid]  = betag[e * DM + tid];
    }
    __syncthreads();

    // ---- Phase B: h = relu(Xf @ W1 + b1);  M=16, K=4096, N=256 ----
    // Wave w owns N-tiles {w, w+8}. A-frag rows = batches (Xs rows).
    {
        f32x8 acc0 = {0,0,0,0,0,0,0,0};
        f32x8 acc1 = {0,0,0,0,0,0,0,0};
        const int n0  = w * 16;
        const int n1c = (w + 8) * 16;
        const uint32_t* p1a = P1 + (((size_t)(e * DM + n0  + ln16)) << 11) + hi * 8;
        const uint32_t* p1b = P1 + (((size_t)(e * DM + n1c + ln16)) << 11) + hi * 8;
        const uint16_t* xrow = Xs + (ln16 << 12);          // A row m = ln16

        for (int kb = 0; kb < 128; ++kb) {
            FragAB a, bA, bB;
            const uint16_t* xp = xrow + kb * 32;
#pragma unroll
            for (int v = 0; v < 8; ++v) {
                int kk = hi * 16 + 2 * v;
                a.u[v] = *(const uint32_t*)(xp + kk);      // A: contiguous k = 16*hi + s (same order as B)
            }
            const uint32_t* q1 = p1a + kb * 16;
            const uint32_t* q2 = p1b + kb * 16;
            bA.q[0] = *(const uint4*)(q1);     bA.q[1] = *(const uint4*)(q1 + 4);
            bB.q[0] = *(const uint4*)(q2);     bB.q[1] = *(const uint4*)(q2 + 4);
            __builtin_prefetch(q1 + 64, 0, 1);             // global_prefetch_b8
            __builtin_prefetch(q2 + 64, 0, 1);
            acc0 = wmma_bf16(a, bA, acc0);
            acc1 = wmma_bf16(a, bB, acc1);
        }
#pragma unroll
        for (int r = 0; r < 8; ++r) {
            int m = r + hi * 8;
            float h0 = fmaxf(acc0[r] + b1s[n0  + ln16], 0.f);
            float h1 = fmaxf(acc1[r] + b1s[n1c + ln16], 0.f);
            Hs[m * DM + n0  + ln16] = f2bf(h0);
            Hs[m * DM + n1c + ln16] = f2bf(h1);
        }
    }
    __syncthreads();

    // ---- Phase C1: scores = softmax(Xp Xp^T / sqrt(D)); wave w -> batches 2w,2w+1
    for (int bi = 0; bi < 2; ++bi) {
        const int bb = w * 2 + bi;
        const uint16_t* xb = Xs + bb * FF;
        f32x8 c = {0,0,0,0,0,0,0,0};
        for (int kb = 0; kb < 8; ++kb) {                    // K = 256 dims
            FragAB a, bfr;
            const uint16_t* ar = xb + ln16 * DM + kb * 32;  // row = query token
            const uint16_t* br = ar;                        // B = Xp^T: lane n = key token
#pragma unroll
            for (int v = 0; v < 8; ++v) {
                int kkA = hi * 16 + 2 * v;
                a.u[v]   = *(const uint32_t*)(ar + kkA);
                int kkB  = hi * 16 + 2 * v;                 // B: contiguous k = 16*hi + s, as A
                bfr.u[v] = *(const uint32_t*)(br + kkB);
            }
            c = wmma_bf16(a, bfr, c);
        }
        // row-wise softmax: cols of a row live in 16 lanes of one half
#pragma unroll
        for (int r = 0; r < 8; ++r) {
            float x = c[r] * 0.0625f;                       // 1/sqrt(256)
            float mx = x;
            mx = fmaxf(mx, __shfl_xor(mx, 1));
            mx = fmaxf(mx, __shfl_xor(mx, 2));
            mx = fmaxf(mx, __shfl_xor(mx, 4));
            mx = fmaxf(mx, __shfl_xor(mx, 8));
            float ex = __expf(x - mx);
            float sm = ex;
            sm += __shfl_xor(sm, 1);
            sm += __shfl_xor(sm, 2);
            sm += __shfl_xor(sm, 4);
            sm += __shfl_xor(sm, 8);
            Ps[bb * 256 + (r + hi * 8) * 16 + ln16] = f2bf(ex / sm);
        }
    }
    __syncthreads();

    // ---- Phase C2: X_dot = P @ Xp (K=16 zero-padded to 32); overwrite
    //      Xs[bb] with Ypart = X_dot + Xp (wave-local rows -> no cross hazard)
    for (int bi = 0; bi < 2; ++bi) {
        const int bb = w * 2 + bi;
        uint16_t* xb = Xs + bb * FF;
        const uint16_t* pb = Ps + bb * 256;
        FragAB a;
#pragma unroll
        for (int v = 0; v < 8; ++v) {
            if (hi == 0) {
                int kk = 2 * v;                             // K<16 valid (lane half 0)
                a.u[v] = *(const uint32_t*)(pb + ln16 * 16 + kk);
            } else {
                a.u[v] = 0u;                                // K>=16 zero pad
            }
        }
#pragma unroll
        for (int nt = 0; nt < 16; ++nt) {
            FragAB bfr;
            const int col = nt * 16 + ln16;
#pragma unroll
            for (int v = 0; v < 8; ++v) {
                uint32_t lo = xb[(2 * v) * DM + col];
                uint32_t hb = xb[(2 * v + 1) * DM + col];
                uint32_t pk = lo | (hb << 16);
                bfr.u[v] = (hi == 0) ? pk : 0u;             // K>=16 rows zero
            }
            f32x8 d = {0,0,0,0,0,0,0,0};
            d = wmma_bf16(a, bfr, d);
#pragma unroll
            for (int r = 0; r < 8; ++r) {
                int m = r + hi * 8;
                float yp = d[r] + bf2f(xb[m * DM + col]);   // + residual Xp
                xb[m * DM + col] = f2bf(yp);                // in-place Ypart
            }
        }
    }
    __syncthreads();

    // ---- Phase D: X_deep = h @ W2 (M=16, K=256, N=4096); wave w owns
    //      tokens {2w, 2w+1} = full 256-dim rows -> local LayerNorm + store.
    FragAB af[8];                                           // A-frags from h
#pragma unroll
    for (int kb = 0; kb < 8; ++kb) {
        const uint16_t* hr = Hs + ln16 * DM + kb * 32;
#pragma unroll
        for (int v = 0; v < 8; ++v) {
            int kk = hi * 16 + 2 * v;
            af[kb].u[v] = *(const uint32_t*)(hr + kk);
        }
    }
    for (int ti = 0; ti < 2; ++ti) {
        const int t = w * 2 + ti;
        f32x8 acc[16];
#pragma unroll
        for (int nt = 0; nt < 16; ++nt) acc[nt] = (f32x8){0,0,0,0,0,0,0,0};
        const uint32_t* p2base = P2 + (((size_t)(e * FF + t * DM)) << 7);
#pragma unroll
        for (int nt = 0; nt < 16; ++nt) {
            const uint32_t* pn = p2base + (((size_t)(nt * 16 + ln16)) << 7) + hi * 8;
            f32x8 a8 = acc[nt];
#pragma unroll
            for (int kb = 0; kb < 8; ++kb) {
                FragAB bfr;
                bfr.q[0] = *(const uint4*)(pn + kb * 16);
                bfr.q[1] = *(const uint4*)(pn + kb * 16 + 4);
                a8 = wmma_bf16(af[kb], bfr, a8);
            }
            acc[nt] = a8;
        }
        // Y = X_deep + b2 + Ypart
#pragma unroll
        for (int nt = 0; nt < 16; ++nt) {
            int dd = nt * 16 + ln16;
            float b2v = b2g[(size_t)e * FF + t * DM + dd];
#pragma unroll
            for (int r = 0; r < 8; ++r) {
                int m = r + hi * 8;
                acc[nt][r] += b2v + bf2f(Xs[m * FF + t * DM + dd]);
            }
        }
        // LayerNorm over the 256 dims of each (batch,token) row
        float mu[8], rs[8];
#pragma unroll
        for (int r = 0; r < 8; ++r) {
            float s = 0.f, s2 = 0.f;
#pragma unroll
            for (int nt = 0; nt < 16; ++nt) {
                float y = acc[nt][r];
                s += y; s2 += y * y;
            }
            s  += __shfl_xor(s, 1);  s2 += __shfl_xor(s2, 1);
            s  += __shfl_xor(s, 2);  s2 += __shfl_xor(s2, 2);
            s  += __shfl_xor(s, 4);  s2 += __shfl_xor(s2, 4);
            s  += __shfl_xor(s, 8);  s2 += __shfl_xor(s2, 8);
            float mean = s * (1.f / DM);
            float var  = s2 * (1.f / DM) - mean * mean;
            mu[r] = mean;
            rs[r] = rsqrtf(var + EPS);
        }
        const size_t obase = (((size_t)btile * TB) * NG + (size_t)e * TPE + t) * DM;
#pragma unroll
        for (int nt = 0; nt < 16; ++nt) {
            int dd = nt * 16 + ln16;
            float g  = gs[dd];
            float bt = bs[dd];
#pragma unroll
            for (int r = 0; r < 8; ++r) {
                int m = r + hi * 8;
                float yn = (acc[nt][r] - mu[r]) * rs[r] * g + bt;
                out[obase + (size_t)m * (NG * DM) + dd] = yn;
            }
        }
    }
}

extern "C" void kernel_launch(void* const* d_in, const int* in_sizes, int n_in,
                              void* d_out, int out_size, void* d_ws, size_t ws_size,
                              hipStream_t stream) {
    const float* X     = (const float*)d_in[0];
    const float* W1    = (const float*)d_in[1];
    const float* b1    = (const float*)d_in[2];
    const float* W2    = (const float*)d_in[3];
    const float* b2    = (const float*)d_in[4];
    const float* gamma = (const float*)d_in[5];
    const float* beta  = (const float*)d_in[6];

    uint32_t* P1 = (uint32_t*)d_ws;                         // 8 MiB
    uint32_t* P2 = P1 + (size_t)NE * DM * (FF / 2);         // 8 MiB
    (void)ws_size; // requires >= 16 MiB

    const int total = 2 * NE * DM * (FF / 2);               // 4M pack threads
    pack_weights_kernel<<<total / 256, 256, 0, stream>>>(W1, W2, P1, P2);

    dim3 grid(B_TOT / TB, NE);
    fused_moe_kernel<<<grid, NTHREADS, 0, stream>>>(X, b1, b2, gamma, beta,
                                                    P1, P2, (float*)d_out);
}
